// FullMultiHeadAttentionTreeLSTMCell_39273180954992
// MI455X (gfx1250) — matrix operand order repaired and checked by reading, not compiled, not run on hardware
//
#include <hip/hip_runtime.h>
#include <stdint.h>

// ---------------- problem constants ----------------
#define K_ARY   4
#define LEVELS  9
#define N_NODES 87381          // (4^9-1)/3
#define HS      256            // X_SIZE == H_SIZE == A_SIZE
#define NHEADS  8
#define AK      32
#define CHUNK   8192           // node chunk -> ~161 MB workspace bound

// ---------------- WMMA types ----------------
typedef __attribute__((ext_vector_type(16))) __bf16 v16bf;
typedef __attribute__((ext_vector_type(8)))  float  v8f;

union FragB { uint4 q[2]; v16bf v; };   // 32B = one lane's 16 bf16 fragment

__device__ __forceinline__ uint16_t f2bf(float f) {       // fp32 -> bf16 RNE
  uint32_t u = __float_as_uint(f);
  return (uint16_t)((u + 0x7FFFu + ((u >> 16) & 1u)) >> 16);
}
__device__ __forceinline__ float sigm(float v) { return 1.0f / (1.0f + __expf(-v)); }

// ---------------- weight pre-pack (fp32 -> bf16, WMMA-fragment-swizzled) ----
// Pack layout (bf16 units): elem = ((nt*8 + kt)*32 + lane)*16 + j
//   Consumer does two contiguous global_load_b128 per fragment (coalesced).
//   B fragment mapping (contiguous k order, same as A): N = nt*16 + (lane&15),
//                                            K = kt*32 + (lane>>4)*16 + j.
__global__ void pack_weights_kernel(uint16_t* __restrict__ dst,
                                    const float* __restrict__ s0, int c0,
                                    const float* __restrict__ s1, int c1,
                                    const float* __restrict__ s2, int c2,
                                    int Ntot) {
  int f = blockIdx.x * 256 + threadIdx.x;
  if (f >= 256 * Ntot) return;
  int j    = f & 15;
  int lane = (f >> 4) & 31;
  int kt   = (f >> 9) & 7;
  int nt   = f >> 12;
  int K = kt * 32 + ((lane >> 4) * 16) + j;   // B-matrix K mapping
  int N = nt * 16 + (lane & 15);
  const float* src; int col, nc;
  if (N < c0)            { src = s0; col = N;           nc = c0; }
  else if (N < c0 + c1)  { src = s1; col = N - c0;      nc = c1; }
  else                   { src = s2; col = N - c0 - c1; nc = c2; }
  dst[f] = f2bf(src[(size_t)K * nc + col]);
}

__global__ void pack_bias_kernel(float* __restrict__ dst,
                                 const float* __restrict__ s0, int c0,
                                 const float* __restrict__ s1, int c1,
                                 const float* __restrict__ s2, int c2) {
  int i = blockIdx.x * 256 + threadIdx.x;
  if (i >= c0 + c1 + c2) return;
  dst[i] = (i < c0) ? s0[i] : (i < c0 + c1) ? s1[i - c0] : s2[i - c0 - c1];
}

// ---------------- bf16 WMMA GEMM: C[M,N] = A[M,256](f32) * Bpack + bias -----
// Block: 256 threads = 8 waves; block tile 64 x 256.
// Each wave owns a 64-row x 32-col strip: per K-step it loads 2 B fragments
// (each feeds 4 WMMAs) and 4 A fragments (each feeds 2 WMMAs) -> 8 WMMA/step,
// 64 WMMA per block pass, no duplicate B loads within the block.
// A staged in LDS as bf16, row stride 264 (=132 dwords): 16 row-lanes map to
// distinct 4-bank groups -> optimal 2-access ds_load_b128 pattern.
__global__ __launch_bounds__(256) void gemm_bf16_kernel(
    const float* __restrict__ A, int M,
    const uint16_t* __restrict__ Bpack,
    const float* __restrict__ biasPack,
    float* __restrict__ C, int ldc) {
  __shared__ uint16_t aLds[64 * 264];
  const int t    = threadIdx.x;
  const int lane = t & 31;
  const int wave = t >> 5;
  const int tileM  = blockIdx.x * 64;
  const int ntBase = blockIdx.y * 16 + wave * 2;   // wave cols [wave*32, +32)

  // stage A tile: 64 x 256 f32 -> bf16 (8192 float2 pairs, coalesced)
  #pragma unroll
  for (int i = 0; i < 32; ++i) {
    int p   = i * 256 + t;
    int row = p >> 7;
    int kp  = p & 127;
    float2 v = make_float2(0.f, 0.f);
    int gRow = tileM + row;
    if (gRow < M) v = *(const float2*)(A + (size_t)gRow * HS + kp * 2);
    uint32_t packed = (uint32_t)f2bf(v.x) | ((uint32_t)f2bf(v.y) << 16);
    *(uint32_t*)&aLds[row * 264 + kp * 2] = packed;
  }
  __syncthreads();

  v8f acc[4][2];
  #pragma unroll
  for (int mf = 0; mf < 4; ++mf)
    #pragma unroll
    for (int f = 0; f < 2; ++f)
      #pragma unroll
      for (int r = 0; r < 8; ++r) acc[mf][f][r] = 0.0f;

  const int m  = lane & 15;
  const int kb = (lane >> 4) * 16;  // A-matrix K map: {kb..kb+15}

  #pragma unroll
  for (int kt = 0; kt < 8; ++kt) {
    FragB bfr[2];
    #pragma unroll
    for (int f = 0; f < 2; ++f) {
      const uint16_t* bp = Bpack + ((((size_t)(ntBase + f) * 8 + kt) * 32 + lane) * 16);
      bfr[f].q[0] = *(const uint4*)(bp);
      bfr[f].q[1] = *(const uint4*)(bp + 8);
    }
    #pragma unroll
    for (int mf = 0; mf < 4; ++mf) {
      FragB af;
      const uint16_t* ap = &aLds[(mf * 16 + m) * 264 + kt * 32 + kb];
      af.q[0] = *(const uint4*)(ap);
      af.q[1] = *(const uint4*)(ap + 8);
      acc[mf][0] = __builtin_amdgcn_wmma_f32_16x16x32_bf16(
          false, af.v, false, bfr[0].v, (short)0, acc[mf][0], false, false);
      acc[mf][1] = __builtin_amdgcn_wmma_f32_16x16x32_bf16(
          false, af.v, false, bfr[1].v, (short)0, acc[mf][1], false, false);
    }
  }

  // epilogue: D layout = 8 VGPRs, lanes 0-15 -> M=r, lanes 16-31 -> M=8+r
  const int rbase = (lane >> 4) * 8;
  #pragma unroll
  for (int f = 0; f < 2; ++f) {
    int col = (ntBase + f) * 16 + (lane & 15);
    float bias = biasPack[col];
    #pragma unroll
    for (int mf = 0; mf < 4; ++mf) {
      #pragma unroll
      for (int r = 0; r < 8; ++r) {
        int row = tileM + mf * 16 + rbase + r;
        if (row < M) C[(size_t)row * ldc + col] = acc[mf][f][r] + bias;
      }
    }
  }
}

// ---------------- attention: one block per node, one wave per head ---------
__global__ __launch_bounds__(256) void attn_kernel(
    const float* __restrict__ Tx,     // [cn,1280], q at col offset 1024
    const float* __restrict__ Tkv,    // [4*cn,512], k at 0, v at 256
    float* __restrict__ Tout) {       // [cn,256]
  int node = blockIdx.x;
  int head = threadIdx.x >> 5;
  int lane = threadIdx.x & 31;
  const float scale = 0.17677669529663687f;   // 1/sqrt(32)
  float q = Tx[(size_t)node * 1280 + 1024 + head * 32 + lane];
  float lg[4];
  #pragma unroll
  for (int k = 0; k < 4; ++k) {
    float kk = Tkv[(size_t)(node * 4 + k) * 512 + head * 32 + lane];
    float d = q * kk;
    #pragma unroll
    for (int off = 16; off > 0; off >>= 1) d += __shfl_xor(d, off, 32);
    lg[k] = d * scale;
  }
  float mx = fmaxf(fmaxf(lg[0], lg[1]), fmaxf(lg[2], lg[3]));
  float e0 = __expf(lg[0] - mx), e1 = __expf(lg[1] - mx);
  float e2 = __expf(lg[2] - mx), e3 = __expf(lg[3] - mx);
  float inv = 1.0f / (e0 + e1 + e2 + e3);
  float w[4] = {e0 * inv, e1 * inv, e2 * inv, e3 * inv};
  float out = 0.0f;
  #pragma unroll
  for (int k = 0; k < 4; ++k)
    out += w[k] * Tkv[(size_t)(node * 4 + k) * 512 + 256 + head * 32 + lane];
  Tout[(size_t)node * 256 + head * 32 + lane] = out;
}

// ---------------- LSTM gate epilogue: one block per node, thread per col ----
__global__ __launch_bounds__(256) void gate_kernel(
    const float* __restrict__ Tx, int ldx,     // xi@0 xo@256 xu@512 xf@768
    const float* __restrict__ Thu,             // [cn,1024] (internal only)
    const float* __restrict__ cChild,          // children c rows (internal only)
    float* __restrict__ hOut, float* __restrict__ cOut,
    int internal) {
  int node = blockIdx.x;
  int col  = threadIdx.x;
  size_t tx = (size_t)node * ldx;
  float i = Tx[tx + col], o = Tx[tx + 256 + col], u = Tx[tx + 512 + col];
  float fc = 0.0f;
  if (internal) {
    size_t th = (size_t)node * 1024;
    i += Thu[th + col]; o += Thu[th + 256 + col]; u += Thu[th + 512 + col];
    float fgate = sigm(Tx[tx + 768 + col] + Thu[th + 768 + col]);
    float cs = 0.0f;
    #pragma unroll
    for (int k = 0; k < 4; ++k) cs += cChild[(size_t)(node * 4 + k) * 256 + col];
    fc = fgate * cs;
  }
  float cv = sigm(i) * tanhf(u) + fc;
  float hv = sigm(o) * tanhf(cv);
  hOut[(size_t)node * 256 + col] = hv;
  cOut[(size_t)node * 256 + col] = cv;
}

// ---------------- host orchestration ----------------
extern "C" void kernel_launch(void* const* d_in, const int* in_sizes, int n_in,
                              void* d_out, int out_size, void* d_ws, size_t ws_size,
                              hipStream_t stream) {
  const float* x      = (const float*)d_in[0];
  const float* W_iou  = (const float*)d_in[1];
  const float* b_iou  = (const float*)d_in[2];
  const float* W_f    = (const float*)d_in[3];
  const float* b_f    = (const float*)d_in[4];
  const float* Wq     = (const float*)d_in[5];
  const float* bq     = (const float*)d_in[6];
  const float* Wk     = (const float*)d_in[7];
  const float* bk     = (const float*)d_in[8];
  const float* Wv     = (const float*)d_in[9];
  const float* bv     = (const float*)d_in[10];
  const float* Wl     = (const float*)d_in[11];
  const float* bl     = (const float*)d_in[12];
  const float* Uiou_w = (const float*)d_in[13];
  const float* Uiou_b = (const float*)d_in[14];
  const float* Uf_w   = (const float*)d_in[15];
  const float* Uf_b   = (const float*)d_in[16];

  float* hG = (float*)d_out;                      // h: [N_NODES,256]
  float* cG = hG + (size_t)N_NODES * HS;          // c: [N_NODES,256]

  // workspace carve-up (~161 MB total)
  char* ws = (char*)d_ws; size_t off = 0;
  auto carve = [&](size_t bytes) { void* p = ws + off; off = (off + bytes + 255) & ~(size_t)255; return p; };
  uint16_t* WB1 = (uint16_t*)carve((size_t)256 * 1280 * 2);  // [W_iou|W_f|Wq]
  uint16_t* WB2 = (uint16_t*)carve((size_t)256 * 512  * 2);  // [Wk|Wv]
  uint16_t* WB3 = (uint16_t*)carve((size_t)256 * 256  * 2);  // Wl
  uint16_t* WB4 = (uint16_t*)carve((size_t)256 * 1024 * 2);  // [Uiou_w|Uf_w]
  float* bias1 = (float*)carve(1280 * 4);
  float* bias2 = (float*)carve(512  * 4);
  float* bias3 = (float*)carve(256  * 4);
  float* bias4 = (float*)carve(1024 * 4);
  float* Tx    = (float*)carve((size_t)CHUNK * 1280 * 4);
  float* Tkv   = (float*)carve((size_t)CHUNK * 4 * 512 * 4);
  float* Tout  = (float*)carve((size_t)CHUNK * 256 * 4);
  float* Thatt = (float*)carve((size_t)CHUNK * 256 * 4);
  float* Thu   = (float*)carve((size_t)CHUNK * 1024 * 4);

  // pack weights (bf16, fragment-swizzled) + biases
  pack_weights_kernel<<<(256 * 1280 + 255) / 256, 256, 0, stream>>>(WB1, W_iou, 768, W_f, 256, Wq, 256, 1280);
  pack_weights_kernel<<<(256 * 512  + 255) / 256, 256, 0, stream>>>(WB2, Wk, 256, Wv, 256, Wv, 0, 512);
  pack_weights_kernel<<<(256 * 256  + 255) / 256, 256, 0, stream>>>(WB3, Wl, 256, Wl, 0, Wl, 0, 256);
  pack_weights_kernel<<<(256 * 1024 + 255) / 256, 256, 0, stream>>>(WB4, Uiou_w, 768, Uf_w, 256, Uf_w, 0, 1024);
  pack_bias_kernel<<<(1280 + 255) / 256, 256, 0, stream>>>(bias1, b_iou, 768, b_f, 256, bq, 256);
  pack_bias_kernel<<<(512  + 255) / 256, 256, 0, stream>>>(bias2, bk, 256, bv, 256, bv, 0);
  pack_bias_kernel<<<(256  + 255) / 256, 256, 0, stream>>>(bias3, bl, 256, bl, 0, bl, 0);
  pack_bias_kernel<<<(1024 + 255) / 256, 256, 0, stream>>>(bias4, Uiou_b, 768, Uf_b, 256, Uf_b, 0);

  int startsArr[LEVELS + 1]; startsArr[0] = 0;
  int npow[LEVELS];
  { int pw = 1; for (int l = 0; l < LEVELS; ++l) { npow[l] = pw; startsArr[l + 1] = startsArr[l] + pw; pw *= 4; } }

  for (int l = LEVELS - 1; l >= 0; --l) {
    int n = npow[l], s0 = startsArr[l];
    bool leaf = (l == LEVELS - 1);
    int childS0 = leaf ? 0 : startsArr[l + 1];
    for (int n0 = 0; n0 < n; n0 += CHUNK) {
      int cn = n - n0; if (cn > CHUNK) cn = CHUNK;
      int Nc = leaf ? 1024 : 1280;              // leaves don't need q columns
      dim3 g1((cn + 63) / 64, Nc / 256);
      gemm_bf16_kernel<<<g1, 256, 0, stream>>>(
          x + (size_t)(s0 + n0) * HS, cn, WB1, bias1, Tx, Nc);
      if (!leaf) {
        dim3 g2((4 * cn + 63) / 64, 2);
        gemm_bf16_kernel<<<g2, 256, 0, stream>>>(
            hG + (size_t)(childS0 + 4 * n0) * HS, 4 * cn, WB2, bias2, Tkv, 512);
        attn_kernel<<<cn, 256, 0, stream>>>(Tx, Tkv, Tout);
        dim3 g3((cn + 63) / 64, 1);
        gemm_bf16_kernel<<<g3, 256, 0, stream>>>(Tout, cn, WB3, bias3, Thatt, 256);
        dim3 g4((cn + 63) / 64, 4);
        gemm_bf16_kernel<<<g4, 256, 0, stream>>>(Thatt, cn, WB4, bias4, Thu, 1024);
      }
      gate_kernel<<<cn, 256, 0, stream>>>(
          Tx, Nc,
          leaf ? nullptr : Thu,
          leaf ? nullptr : (cG + (size_t)(childS0 + 4 * n0) * HS),
          hG + (size_t)(s0 + n0) * HS,
          cG + (size_t)(s0 + n0) * HS,
          leaf ? 0 : 1);
    }
  }
  (void)in_sizes; (void)n_in; (void)out_size; (void)ws_size;
}
